// GatedRNNCell_36567351558875
// MI455X (gfx1250) — matrix operand order repaired and checked by reading, not compiled, not run on hardware
//
#include <hip/hip_runtime.h>

// ---------------------------------------------------------------------------
// GatedRNNCell on MI455X (gfx1250, wave32, WMMA + async global->LDS).
// f16 operands + f32 accumulation via v_wmma_f32_16x16x32_f16.
// Double-buffered LDS fed by global_load_async_to_lds_b128 (ASYNCcnt).
// B=4096, IN=1024, H=2048, CTX=1024.
// ---------------------------------------------------------------------------

typedef _Float16 half8 __attribute__((ext_vector_type(8)));
typedef _Float16 v16h  __attribute__((ext_vector_type(16)));
typedef float    v8f   __attribute__((ext_vector_type(8)));
typedef float    f4    __attribute__((ext_vector_type(4)));
typedef _Float16 h4    __attribute__((ext_vector_type(4)));

#define GB   4096  // batch (M)
#define GH   2048  // hidden (N)
#define GIN  1024
#define GCTX 1024

// ----------------------------- f32 -> f16 conversion -----------------------
__global__ __launch_bounds__(256) void f32_to_f16_kernel(
    const f4* __restrict__ in, h4* __restrict__ out, int n4) {
  int i = blockIdx.x * 256 + threadIdx.x;
  int stride = gridDim.x * 256;
  for (; i < n4; i += stride) out[i] = __builtin_convertvector(in[i], h4);
}

__device__ __forceinline__ float fast_sigmoid(float x) {
  // 1/(1+e^-x) with hardware v_rcp_f32 (1 ulp) instead of IEEE divide.
  return __builtin_amdgcn_rcpf(1.0f + __expf(-x));
}
__device__ __forceinline__ float fast_tanh(float x) {
  // tanh(x) = 1 - 2/(e^{2x}+1); saturates correctly at +-1.
  return 1.0f - 2.0f * __builtin_amdgcn_rcpf(__expf(2.0f * x) + 1.0f);
}

// ----------------------------- fused GEMM ----------------------------------
// D[M=4096][N=2048] = sum_t A_t (MxK_t f16 row-major) @ B_t (NxK_t f16)^T
// K sizes are template constants so all pipeline control stays uniform (SALU).
// EP: 0: outF = acc + bias[n]
//     1: outF = sigmoid(acc + bias[n])
//     2: outH = f16( sigmoid(acc + bias[n]) * sprev )
//     3: outF = (1-z)*sprev + z*tanh(acc + p)
template <int EP, int K0v, int K1v, int K2v>
__global__ __launch_bounds__(256) void gemm3_f16(
    const _Float16* __restrict__ A0, const _Float16* __restrict__ B0,
    const _Float16* __restrict__ A1, const _Float16* __restrict__ B1,
    const _Float16* __restrict__ A2, const _Float16* __restrict__ B2,
    const float* __restrict__ bias,
    const float* __restrict__ sprev,
    const float* __restrict__ pbuf,
    const float* __restrict__ zbuf,
    float* __restrict__ outF,
    _Float16* __restrict__ outH)
{
  constexpr int BK  = 64;   // K per LDS tile (2 WMMA substeps)
  constexpr int LDR = 72;   // padded LDS row stride in f16 (144 B, 16B aligned)
  constexpr int S0 = K0v / BK, S1 = K1v / BK, S2 = K2v / BK;
  constexpr int TOTAL = S0 + S1 + S2;

  // Double-buffered tiles: 2 * (128*72 + 128*72) * 2B = 72 KB.
  __shared__ _Float16 Asm[2][128 * LDR];
  __shared__ _Float16 Bsm[2][128 * LDR];

  const int tid    = threadIdx.x;
  const int lane   = tid & 31;
  const int wave   = tid >> 5;
  const int waveM  = wave & 3;        // 4 waves along M -> 32 rows each
  const int waveN  = wave >> 2;       // 2 waves along N -> 64 cols each
  const int laneM  = lane & 15;
  const int laneHi = lane >> 4;

  const int ldrow = tid >> 3;         // 0..31 (4 passes cover 128 rows)
  const int ldcol = (tid & 7) * 8;    // 0..56

  const int bm = blockIdx.y * 128;
  const int bn = blockIdx.x * 128;

  // Issue the 8 async 16B copies (4 A-rows + 4 B-rows) for tile `step` -> buf.
  // Segment selection is uniform scalar compares against constexpr bounds.
  auto issue_tile = [&](int step, int buf) {
    const _Float16* Ag;
    const _Float16* Bg;
    int K, k0;
    if constexpr (S1 == 0) {
      Ag = A0; Bg = B0; K = K0v; k0 = step * BK;
    } else if constexpr (S2 == 0) {
      if (step < S0) { Ag = A0; Bg = B0; K = K0v; k0 = step * BK; }
      else           { Ag = A1; Bg = B1; K = K1v; k0 = (step - S0) * BK; }
    } else {
      if (step < S0)           { Ag = A0; Bg = B0; K = K0v; k0 = step * BK; }
      else if (step < S0 + S1) { Ag = A1; Bg = B1; K = K1v; k0 = (step - S0) * BK; }
      else                     { Ag = A2; Bg = B2; K = K2v; k0 = (step - S0 - S1) * BK; }
    }
    const unsigned ldsA = (unsigned)(uintptr_t)&Asm[buf][0];
    const unsigned ldsB = (unsigned)(uintptr_t)&Bsm[buf][0];
#pragma unroll
    for (int p = 0; p < 4; ++p) {
      const int r = ldrow + 32 * p;
      const unsigned lo = (unsigned)(r * LDR + ldcol) * 2u;
      const unsigned ga = (unsigned)(((size_t)(bm + r) * K + (k0 + ldcol)) * 2u);
      const unsigned gb = (unsigned)(((size_t)(bn + r) * K + (k0 + ldcol)) * 2u);
      unsigned la = ldsA + lo, lb = ldsB + lo;
      asm volatile("global_load_async_to_lds_b128 %0, %1, %2"
                   :: "v"(la), "v"(ga), "s"(Ag) : "memory");
      asm volatile("global_load_async_to_lds_b128 %0, %1, %2"
                   :: "v"(lb), "v"(gb), "s"(Bg) : "memory");
    }
  };

  v8f acc[2][4] = {};

  issue_tile(0, 0);
  for (int step = 0; step < TOTAL; ++step) {
    const int cb = step & 1;
    // Branch-free tail: last iteration re-issues tile 0 into the dead buffer
    // (8 extra 16B loads; drained by s_endpgm's implicit wait-idle).
    const int nxt = (step + 1 < TOTAL) ? (step + 1) : 0;
    issue_tile(nxt, cb ^ 1);
    // 16 outstanding: wait until the current tile's 8 are done (in-order);
    // the next tile's 8 stay in flight during compute.
    asm volatile("s_wait_asynccnt 0x8" ::: "memory");
    __syncthreads();

#pragma unroll
    for (int ks = 0; ks < BK; ks += 32) {
      // A (16x32 MxK): lane holds row laneM; halves 0..7 = K[8*laneHi..+7],
      //                halves 8..15 = K[16+8*laneHi..+7]
      v16h af[2];
#pragma unroll
      for (int i = 0; i < 2; ++i) {
        const _Float16* pA =
            &Asm[cb][(waveM * 32 + i * 16 + laneM) * LDR + ks + laneHi * 8];
        half8 lo = *(const half8*)pA;
        half8 hi = *(const half8*)(pA + 16);
        af[i] = __builtin_shufflevector(lo, hi, 0, 1, 2, 3, 4, 5, 6, 7,
                                                8, 9, 10, 11, 12, 13, 14, 15);
      }
      // B (32x16 KxN): lane holds column laneM (= weight row),
      //                halves 0..7 = K[8*laneHi..+7], halves 8..15 = K[16+8*laneHi..+7]
      v16h bf[4];
#pragma unroll
      for (int j = 0; j < 4; ++j) {
        const _Float16* pB =
            &Bsm[cb][(waveN * 64 + j * 16 + laneM) * LDR + ks + laneHi * 8];
        half8 lo = *(const half8*)pB;
        half8 hi = *(const half8*)(pB + 16);
        bf[j] = __builtin_shufflevector(lo, hi, 0, 1, 2, 3, 4, 5, 6, 7,
                                                8, 9, 10, 11, 12, 13, 14, 15);
      }
#pragma unroll
      for (int i = 0; i < 2; ++i)
#pragma unroll
        for (int j = 0; j < 4; ++j)
          acc[i][j] = __builtin_amdgcn_wmma_f32_16x16x32_f16(
              false, af[i], false, bf[j], (short)0, acc[i][j], false, false);
    }
    __syncthreads();
  }

  // ---- epilogue: C/D layout: lane covers col laneM, rows v + 8*laneHi ----
#pragma unroll
  for (int i = 0; i < 2; ++i) {
#pragma unroll
    for (int j = 0; j < 4; ++j) {
      const int n     = bn + waveN * 64 + j * 16 + laneM;
      const int mBase = bm + waveM * 32 + i * 16 + laneHi * 8;
      const float bn_f = (EP == 3) ? 0.0f : bias[n];
#pragma unroll
      for (int v = 0; v < 8; ++v) {
        const size_t idx = (size_t)(mBase + v) * GH + n;
        const float a = acc[i][j][v];
        if (EP == 0) {
          outF[idx] = a + bn_f;
        } else if (EP == 1) {
          outF[idx] = fast_sigmoid(a + bn_f);
        } else if (EP == 2) {
          outH[idx] = (_Float16)(fast_sigmoid(a + bn_f) * sprev[idx]);
        } else {
          const float cand = fast_tanh(a + pbuf[idx]);  // p already has bias b
          const float zz   = zbuf[idx];
          const float sp   = sprev[idx];
          outF[idx] = (1.0f - zz) * sp + zz * cand;
        }
      }
    }
  }
}

// ----------------------------- launcher ------------------------------------
extern "C" void kernel_launch(void* const* d_in, const int* in_sizes, int n_in,
                              void* d_out, int out_size, void* d_ws, size_t ws_size,
                              hipStream_t stream) {
  const float* x  = (const float*)d_in[0];
  const float* s  = (const float*)d_in[1];
  const float* c  = (const float*)d_in[2];
  const float* W  = (const float*)d_in[3];
  const float* Wz = (const float*)d_in[4];
  const float* Wr = (const float*)d_in[5];
  const float* U  = (const float*)d_in[6];
  const float* Uz = (const float*)d_in[7];
  const float* Ur = (const float*)d_in[8];
  const float* C  = (const float*)d_in[9];
  const float* Cz = (const float*)d_in[10];
  const float* Cr = (const float*)d_in[11];
  const float* b  = (const float*)d_in[12];
  const float* bz = (const float*)d_in[13];
  const float* br = (const float*)d_in[14];
  float* out = (float*)d_out;

  // Workspace carve-up (~168 MB total).
  char* ws = (char*)d_ws;
  size_t off = 0;
  auto alloc_h = [&](size_t elems) -> _Float16* {
    _Float16* p = (_Float16*)(ws + off);
    off += ((elems * 2 + 255) / 256) * 256;
    return p;
  };
  auto alloc_f = [&](size_t elems) -> float* {
    float* p = (float*)(ws + off);
    off += ((elems * 4 + 255) / 256) * 256;
    return p;
  };

  _Float16* x16  = alloc_h((size_t)GB * GIN);
  _Float16* s16  = alloc_h((size_t)GB * GH);
  _Float16* c16  = alloc_h((size_t)GB * GCTX);
  _Float16* W16  = alloc_h((size_t)GH * GIN);
  _Float16* Wz16 = alloc_h((size_t)GH * GIN);
  _Float16* Wr16 = alloc_h((size_t)GH * GIN);
  _Float16* U16  = alloc_h((size_t)GH * GH);
  _Float16* Uz16 = alloc_h((size_t)GH * GH);
  _Float16* Ur16 = alloc_h((size_t)GH * GH);
  _Float16* C16  = alloc_h((size_t)GH * GCTX);
  _Float16* Cz16 = alloc_h((size_t)GH * GCTX);
  _Float16* Cr16 = alloc_h((size_t)GH * GCTX);
  _Float16* h16  = alloc_h((size_t)GB * GH);
  float*    z32  = alloc_f((size_t)GB * GH);
  float*    p32  = alloc_f((size_t)GB * GH);

  auto conv = [&](const float* src, _Float16* dst, int n) {
    int n4 = n / 4;
    int blocks = (n4 + 1023) / 1024;
    f32_to_f16_kernel<<<blocks, 256, 0, stream>>>((const f4*)src, (h4*)dst, n4);
  };
  conv(x,  x16,  GB * GIN);
  conv(s,  s16,  GB * GH);
  conv(c,  c16,  GB * GCTX);
  conv(W,  W16,  GH * GIN);
  conv(Wz, Wz16, GH * GIN);
  conv(Wr, Wr16, GH * GIN);
  conv(U,  U16,  GH * GH);
  conv(Uz, Uz16, GH * GH);
  conv(Ur, Ur16, GH * GH);
  conv(C,  C16,  GH * GCTX);
  conv(Cz, Cz16, GH * GCTX);
  conv(Cr, Cr16, GH * GCTX);

  dim3 grid(GH / 128, GB / 128);  // (16, 32)
  dim3 block(256);

  // h = sigmoid(x Wr^T + s Ur^T + c Cr^T + br) * s_prev   (f16)
  gemm3_f16<2, GIN, GH, GCTX><<<grid, block, 0, stream>>>(
      x16, Wr16, s16, Ur16, c16, Cr16,
      br, s, nullptr, nullptr, nullptr, h16);

  // z = sigmoid(x Wz^T + s Uz^T + c Cz^T + bz)            (f32)
  gemm3_f16<1, GIN, GH, GCTX><<<grid, block, 0, stream>>>(
      x16, Wz16, s16, Uz16, c16, Cz16,
      bz, nullptr, nullptr, nullptr, z32, nullptr);

  // p = x W^T + c C^T + b                                  (f32)
  gemm3_f16<0, GIN, GCTX, 0><<<grid, block, 0, stream>>>(
      x16, W16, c16, C16, nullptr, nullptr,
      b, nullptr, nullptr, nullptr, p32, nullptr);

  // out = (1-z)*s + z*tanh(h U^T + p)
  gemm3_f16<3, GH, 0, 0><<<grid, block, 0, stream>>>(
      h16, U16, nullptr, nullptr, nullptr, nullptr,
      nullptr, s, p32, z32, out, nullptr);
}
